// SwinTransformerBlock_21784074125561
// MI455X (gfx1250) — matrix operand order repaired and checked by reading, not compiled, not run on hardware
//
#include <hip/hip_runtime.h>
#include <hip/hip_bf16.h>

typedef __bf16 v16bf __attribute__((ext_vector_type(16)));
typedef float  v8f   __attribute__((ext_vector_type(8)));

#define BQ    64
#define HDIM  56
#define CC    192
#define NHEAD 6
#define HD    32
#define WS    7
#define SSH   3
#define NTOK  49
#define NWIN  64
#define BWIN  (BQ * NWIN)      // 4096
#define MROWS (BWIN * NTOK)    // 200704
#define HID   768
#define C3    (3 * CC)         // 576
#define SCALE 0.1767766952966369f
#define NEGBIG -3.0e38f

// ---------------- WMMA helpers ----------------
__device__ __forceinline__ v8f v8f_zero() {
  v8f z = {0.f,0.f,0.f,0.f,0.f,0.f,0.f,0.f};
  return z;
}

__device__ __forceinline__ v8f wmma_bf16(v16bf a, v16bf b, v8f c) {
  return __builtin_amdgcn_wmma_f32_16x16x32_bf16(false, a, false, b, (short)0, c, false, false);
}

// A-fragment (16x32 bf16), row-major A with leading dim ldk.
// lane 0-15: row = lane, K = [0..15]; lane 16-31: row = lane-16, K = [16..31]
__device__ __forceinline__ v16bf load_a_frag(const __bf16* A, int ldk, int row0, int k0, int lane) {
  const __bf16* p = A + (size_t)(row0 + (lane & 15)) * ldk + k0 + ((lane >> 4) * 16);
  v16bf a;
#pragma unroll
  for (int e = 0; e < 8; ++e) a[e] = p[e];
#pragma unroll
  for (int e = 0; e < 8; ++e) a[8 + e] = p[8 + e];
  return a;
}

// B-fragment (32x16 bf16) from TRANSPOSED weights WT[N, K]:
// lane 0-15: col = lane, K = [0..15]; lane 16-31: col = lane-16, K = [16..31]
__device__ __forceinline__ v16bf load_bT_frag(const __bf16* WT, int ldk, int col0, int k0, int lane) {
  const __bf16* p = WT + (size_t)(col0 + (lane & 15)) * ldk + k0 + ((lane >> 4) * 16);
  v16bf b;
#pragma unroll
  for (int e = 0; e < 16; ++e) b[e] = p[e];
  return b;
}

// ---------------- weight transpose+convert: W[K,N] f32 -> WT[N,K] bf16 ----------------
__global__ void wt_convert_kernel(const float* __restrict__ W, __bf16* __restrict__ WT,
                                  int K, int Nout) {
  int idx = blockIdx.x * 256 + threadIdx.x;
  if (idx >= K * Nout) return;
  int k = idx / Nout, n = idx - k * Nout;
  WT[(size_t)n * K + k] = (__bf16)W[idx];
}

// ---------------- precompute bias+mask table: T[cls(4)][head(6)][ti(64)][tj(64)] f32 ----------
// cls = (windowIsBottomEdge?2:0) | (windowIsRightEdge?1:0); padding rows/cols = NEGBIG
__global__ void biasmask_kernel(const float* __restrict__ rpb, float* __restrict__ T) {
  int idx = blockIdx.x * 256 + threadIdx.x;
  if (idx >= 4 * NHEAD * 64 * 64) return;
  int tj   = idx & 63;
  int ti   = (idx >> 6) & 63;
  int head = (idx >> 12) % NHEAD;
  int cls  = idx / (NHEAD * 4096);
  float v;
  if (ti < NTOK && tj < NTOK) {
    int yi = ti / WS, xi = ti - yi * WS;
    int yj = tj / WS, xj = tj - yj * WS;
    int rel = (yi - yj + WS - 1) * (2 * WS - 1) + (xi - xj + WS - 1);
    v = rpb[rel * NHEAD + head];
    int wy = (cls & 2) ? 7 : 0, wx = (cls & 1) ? 7 : 0;
    int hi = wy * WS + yi, wi = wx * WS + xi;
    int hj = wy * WS + yj, wj = wx * WS + xj;
    int ci = (hi < 49 ? 0 : (hi < 53 ? 1 : 2)) * 3 + (wi < 49 ? 0 : (wi < 53 ? 1 : 2));
    int cj = (hj < 49 ? 0 : (hj < 53 ? 1 : 2)) * 3 + (wj < 49 ? 0 : (wj < 53 ? 1 : 2));
    if (ci != cj) v -= 100.0f;
  } else {
    v = NEGBIG;
  }
  T[idx] = v;
}

// ---------------- LayerNorm helpers ----------------
__device__ __forceinline__ float wave_sum32(float v) {
#pragma unroll
  for (int off = 16; off >= 1; off >>= 1) v += __shfl_xor(v, off, 32);
  return v;
}

// LN1 + cyclic shift + window partition.  h[row(win,tok), C] bf16
__global__ void ln1_window_kernel(const float* __restrict__ x, const float* __restrict__ w,
                                  const float* __restrict__ b, __bf16* __restrict__ h) {
  int row  = blockIdx.x * 8 + (threadIdx.x >> 5);
  int lane = threadIdx.x & 31;
  int win = row / NTOK, tok = row - win * NTOK;
  int bi = win >> 6, wv = win & 63;
  int wy = wv >> 3, wx = wv & 7;
  int ty = tok / WS, tx = tok - ty * WS;
  int hp = wy * WS + ty + SSH; if (hp >= HDIM) hp -= HDIM;
  int wp = wx * WS + tx + SSH; if (wp >= HDIM) wp -= HDIM;
  const float* xr = x + (((size_t)bi * (HDIM * HDIM)) + hp * HDIM + wp) * CC;

  float v[6];
  float s = 0.f;
#pragma unroll
  for (int i = 0; i < 6; ++i) { v[i] = xr[lane + 32 * i]; s += v[i]; }
  float mu = wave_sum32(s) * (1.0f / CC);
  float var = 0.f;
#pragma unroll
  for (int i = 0; i < 6; ++i) { float d = v[i] - mu; var += d * d; }
  var = wave_sum32(var) * (1.0f / CC);
  float rs = rsqrtf(var + 1e-5f);
  __bf16* hr = h + (size_t)row * CC;
#pragma unroll
  for (int i = 0; i < 6; ++i) {
    int c = lane + 32 * i;
    hr[c] = (__bf16)((v[i] - mu) * rs * w[c] + b[c]);
  }
}

// LN2 on natural-order rows, f32 in -> bf16 out
__global__ void ln2_kernel(const float* __restrict__ x, const float* __restrict__ w,
                           const float* __restrict__ b, __bf16* __restrict__ h) {
  int row  = blockIdx.x * 8 + (threadIdx.x >> 5);
  int lane = threadIdx.x & 31;
  const float* xr = x + (size_t)row * CC;
  float v[6];
  float s = 0.f;
#pragma unroll
  for (int i = 0; i < 6; ++i) { v[i] = xr[lane + 32 * i]; s += v[i]; }
  float mu = wave_sum32(s) * (1.0f / CC);
  float var = 0.f;
#pragma unroll
  for (int i = 0; i < 6; ++i) { float d = v[i] - mu; var += d * d; }
  var = wave_sum32(var) * (1.0f / CC);
  float rs = rsqrtf(var + 1e-5f);
  __bf16* hr = h + (size_t)row * CC;
#pragma unroll
  for (int i = 0; i < 6; ++i) {
    int c = lane + 32 * i;
    hr[c] = (__bf16)((v[i] - mu) * rs * w[c] + b[c]);
  }
}

// ---------------- generic WMMA GEMM: out = A[M,K] @ WT[N,K]^T + bias ----------------
// MODE 0: bf16 out;  MODE 1: GELU(exact) -> bf16 out
// MODE 2: + residual (natural rows) -> f32 out
// MODE 3: + residual at window-reverse/unshift scatter position -> f32 out
template <int MODE>
__global__ void gemm_bf16_kernel(const __bf16* __restrict__ A, const __bf16* __restrict__ WT,
                                 const float* __restrict__ bias, __bf16* __restrict__ out_bf,
                                 float* __restrict__ out_f32, const float* __restrict__ res,
                                 int K, int Nout) {
  int m0 = blockIdx.x * 128;
  int n0 = blockIdx.y * 64;
  int wave = threadIdx.x >> 5;
  int lane = threadIdx.x & 31;
  int rowBase = m0 + (wave >> 1) * 32;
  int colBase = n0 + (wave & 1) * 32;

  v8f acc[2][2];
#pragma unroll
  for (int i = 0; i < 2; ++i)
#pragma unroll
    for (int j = 0; j < 2; ++j) acc[i][j] = v8f_zero();

  for (int k0 = 0; k0 < K; k0 += 32) {
    v16bf af[2], bf[2];
#pragma unroll
    for (int i = 0; i < 2; ++i) af[i] = load_a_frag(A, K, rowBase + i * 16, k0, lane);
#pragma unroll
    for (int j = 0; j < 2; ++j) bf[j] = load_bT_frag(WT, K, colBase + j * 16, k0, lane);
#pragma unroll
    for (int i = 0; i < 2; ++i)
#pragma unroll
      for (int j = 0; j < 2; ++j) acc[i][j] = wmma_bf16(af[i], bf[j], acc[i][j]);
  }

#pragma unroll
  for (int i = 0; i < 2; ++i) {
#pragma unroll
    for (int j = 0; j < 2; ++j) {
#pragma unroll
      for (int r = 0; r < 8; ++r) {
        int row = rowBase + i * 16 + r + 8 * (lane >> 4);
        int col = colBase + j * 16 + (lane & 15);
        float v = acc[i][j][r] + bias[col];
        if (MODE == 0) {
          out_bf[(size_t)row * Nout + col] = (__bf16)v;
        } else if (MODE == 1) {
          float g = 0.5f * v * (1.0f + erff(v * 0.70710678118654752f));
          out_bf[(size_t)row * Nout + col] = (__bf16)g;
        } else if (MODE == 2) {
          size_t o = (size_t)row * Nout + col;
          out_f32[o] = res[o] + v;
        } else {
          int win = row / NTOK, tok = row - win * NTOK;
          int bi = win >> 6, wv = win & 63;
          int wy = wv >> 3, wx = wv & 7;
          int ty = tok / WS, tx = tok - ty * WS;
          int hp = wy * WS + ty + SSH; if (hp >= HDIM) hp -= HDIM;
          int wp = wx * WS + tx + SSH; if (wp >= HDIM) wp -= HDIM;
          size_t o = (((size_t)bi * (HDIM * HDIM)) + hp * HDIM + wp) * (size_t)Nout + col;
          out_f32[o] = res[o] + v;
        }
      }
    }
  }
}

// ---------------- attention: one wave per (window, head), streaming row-strips ----------------
__global__ void attn_kernel(const __bf16* __restrict__ qkv, const float* __restrict__ T,
                            __bf16* __restrict__ o) {
  __shared__ __bf16 Pb[4][64 * 64];   // 32 KB: per-wave P matrix (softmax output)
  int wave = threadIdx.x >> 5;
  int lane = threadIdx.x & 31;
  int gid  = blockIdx.x * 4 + wave;   // window-head id, < 24576
  int win  = gid / NHEAD;
  int head = gid - win * NHEAD;
  const __bf16* base = qkv + (size_t)win * NTOK * C3 + head * HD;
  int wv = win & 63;
  int wy = wv >> 3, wx = wv & 7;
  int cls = ((wy == 7) ? 2 : 0) | ((wx == 7) ? 1 : 0);
  const float* Tb = T + ((size_t)(cls * NHEAD + head) << 12) + (lane & 15);
  __bf16* P = &Pb[wave][0];

  // ---- K^T as B-fragments: 4 column tiles (j), K-dim = head-dim (32) ----
  v16bf kf[4];
#pragma unroll
  for (int ct = 0; ct < 4; ++ct) {
    int j  = ct * 16 + (lane & 15);
    int db = (lane >> 4) * 16;
    if (j < NTOK) {
      const __bf16* p = base + CC + (size_t)j * C3 + db;
#pragma unroll
      for (int e = 0; e < 16; ++e) kf[ct][e] = p[e];
    } else {
#pragma unroll
      for (int e = 0; e < 16; ++e) kf[ct][e] = (__bf16)0.0f;
    }
  }

  // ---- streaming: per 16-row strip compute S, softmax, store P; accumulators die per strip ----
#pragma unroll 1
  for (int rt = 0; rt < 4; ++rt) {
    int tok = rt * 16 + (lane & 15);
    int dh  = (lane >> 4) * 16;
    v16bf qa;
    if (tok < NTOK) {
      const __bf16* p = base + (size_t)tok * C3 + dh;
#pragma unroll
      for (int e = 0; e < 8; ++e) { qa[e] = p[e]; qa[8 + e] = p[8 + e]; }
    } else {
#pragma unroll
      for (int e = 0; e < 16; ++e) qa[e] = (__bf16)0.0f;
    }
    v8f s4[4];
#pragma unroll
    for (int ct = 0; ct < 4; ++ct) s4[ct] = wmma_bf16(qa, kf[ct], v8f_zero());

    // scale + precomputed bias/mask + softmax on this strip
#pragma unroll
    for (int r = 0; r < 8; ++r) {
      int ti = rt * 16 + r + 8 * (lane >> 4);
      const float* tr = Tb + ti * 64;
      float vals[4];
      float m = NEGBIG;
#pragma unroll
      for (int ct = 0; ct < 4; ++ct) {
        float v = s4[ct][r] * SCALE + tr[ct * 16];
        vals[ct] = v;
        m = fmaxf(m, v);
      }
#pragma unroll
      for (int off = 8; off >= 1; off >>= 1) m = fmaxf(m, __shfl_xor(m, off, 32));
      float ex[4];
      float s = 0.f;
#pragma unroll
      for (int ct = 0; ct < 4; ++ct) {
        float e = (vals[ct] <= -1.0e37f) ? 0.0f : __expf(vals[ct] - m);
        ex[ct] = e; s += e;
      }
#pragma unroll
      for (int off = 8; off >= 1; off >>= 1) s += __shfl_xor(s, off, 32);
      float inv = 1.0f / s;
#pragma unroll
      for (int ct = 0; ct < 4; ++ct)
        P[ti * 64 + ct * 16 + (lane & 15)] = (__bf16)(ex[ct] * inv);
    }
  }

  // ---- V as B-fragments: cols = head-dim (2 tiles), K-dim = tokens (2 chunks of 32) ----
  v16bf vf[2][2];
#pragma unroll
  for (int kc = 0; kc < 2; ++kc) {
#pragma unroll
    for (int cd = 0; cd < 2; ++cd) {
      int d  = cd * 16 + (lane & 15);
      int kb = (lane >> 4) * 16;
#pragma unroll
      for (int e = 0; e < 16; ++e) {
        int j = kc * 32 + kb + e;
        vf[kc][cd][e] = (j < NTOK) ? base[2 * CC + (size_t)j * C3 + d] : (__bf16)0.0f;
      }
    }
  }

  // ---- O = P @ V, one row-strip at a time; store windows as [win*NTOK + tok, C] bf16 ----
#pragma unroll 1
  for (int rt = 0; rt < 4; ++rt) {
    v8f oa[2] = {v8f_zero(), v8f_zero()};
#pragma unroll
    for (int kc = 0; kc < 2; ++kc) {
      const __bf16* pp = P + (rt * 16 + (lane & 15)) * 64 + kc * 32 + ((lane >> 4) * 16);
      v16bf pa;
#pragma unroll
      for (int e = 0; e < 8; ++e) { pa[e] = pp[e]; pa[8 + e] = pp[8 + e]; }
#pragma unroll
      for (int cd = 0; cd < 2; ++cd) oa[cd] = wmma_bf16(pa, vf[kc][cd], oa[cd]);
    }
#pragma unroll
    for (int cd = 0; cd < 2; ++cd) {
#pragma unroll
      for (int r = 0; r < 8; ++r) {
        int tok = rt * 16 + r + 8 * (lane >> 4);
        if (tok < NTOK) {
          o[((size_t)win * NTOK + tok) * CC + head * HD + cd * 16 + (lane & 15)] =
              (__bf16)oa[cd][r];
        }
      }
    }
  }
}

// ---------------- launcher ----------------
extern "C" void kernel_launch(void* const* d_in, const int* in_sizes, int n_in,
                              void* d_out, int out_size, void* d_ws, size_t ws_size,
                              hipStream_t stream) {
  const float* x       = (const float*)d_in[0];
  const float* norm1_w = (const float*)d_in[1];
  const float* norm1_b = (const float*)d_in[2];
  const float* qkv_w   = (const float*)d_in[3];
  const float* qkv_b   = (const float*)d_in[4];
  const float* rpb     = (const float*)d_in[5];
  const float* proj_w  = (const float*)d_in[6];
  const float* proj_b  = (const float*)d_in[7];
  const float* norm2_w = (const float*)d_in[8];
  const float* norm2_b = (const float*)d_in[9];
  const float* fc1_w   = (const float*)d_in[10];
  const float* fc1_b   = (const float*)d_in[11];
  const float* fc2_w   = (const float*)d_in[12];
  const float* fc2_b   = (const float*)d_in[13];
  float* out = (float*)d_out;

  char* p = (char*)d_ws;
  const size_t X_BYTES   = (size_t)MROWS * CC * 4;     // 154,140,672
  const size_t H_BYTES   = (size_t)MROWS * CC * 2;     //  77,070,336
  const size_t QKV_BYTES = (size_t)MROWS * C3 * 2;     // 231,211,008
  const size_t WT_BYTES  = ((size_t)C3 * CC + (size_t)CC * CC +
                            (size_t)CC * HID + (size_t)HID * CC) * 2;  // 884,736

  float*  ws_x   = (float*)p;                      // x after attn residual
  __bf16* ws_h   = (__bf16*)(p + X_BYTES);         // LN output (reused for LN2)
  __bf16* ws_qkv = (__bf16*)(p + X_BYTES + H_BYTES);
  __bf16* ws_o   = (__bf16*)(p + X_BYTES + H_BYTES + QKV_BYTES);
  __bf16* ws_mlp = ws_qkv;                         // FC1 out aliases dead qkv+o (exact fit)
  char*   wbase  = p + X_BYTES + H_BYTES + QKV_BYTES + H_BYTES;
  __bf16* wqkvT  = (__bf16*)wbase;                              // [576,192]
  __bf16* wprojT = (__bf16*)(wbase + (size_t)C3 * CC * 2);      // [192,192]
  __bf16* wfc1T  = (__bf16*)(wbase + ((size_t)C3 * CC + (size_t)CC * CC) * 2);          // [768,192]
  __bf16* wfc2T  = (__bf16*)(wbase + ((size_t)C3 * CC + (size_t)CC * CC + (size_t)CC * HID) * 2); // [192,768]
  float*  ws_T   = (float*)(wbase + WT_BYTES);     // bias+mask table [4][6][64][64]

  // 1) weight transpose+convert to bf16; bias+mask table
  wt_convert_kernel<<<(CC * C3 + 255) / 256, 256, 0, stream>>>(qkv_w, wqkvT, CC, C3);
  wt_convert_kernel<<<(CC * CC + 255) / 256, 256, 0, stream>>>(proj_w, wprojT, CC, CC);
  wt_convert_kernel<<<(CC * HID + 255) / 256, 256, 0, stream>>>(fc1_w, wfc1T, CC, HID);
  wt_convert_kernel<<<(HID * CC + 255) / 256, 256, 0, stream>>>(fc2_w, wfc2T, HID, CC);
  biasmask_kernel<<<(4 * NHEAD * 64 * 64) / 256, 256, 0, stream>>>(rpb, ws_T);

  // 2) LN1 + shift + window partition
  ln1_window_kernel<<<MROWS / 8, 256, 0, stream>>>(x, norm1_w, norm1_b, ws_h);

  // 3) QKV GEMM  [200704,192] x [192,576]
  gemm_bf16_kernel<0><<<dim3(MROWS / 128, C3 / 64), 256, 0, stream>>>(
      ws_h, wqkvT, qkv_b, ws_qkv, nullptr, nullptr, CC, C3);

  // 4) windowed attention (one wave per window-head)
  attn_kernel<<<(BWIN * NHEAD) / 4, 128, 0, stream>>>(ws_qkv, ws_T, ws_o);

  // 5) proj GEMM + window-reverse/unshift scatter + residual -> ws_x (f32)
  gemm_bf16_kernel<3><<<dim3(MROWS / 128, CC / 64), 256, 0, stream>>>(
      ws_o, wprojT, proj_b, nullptr, ws_x, x, CC, CC);

  // 6) LN2
  ln2_kernel<<<MROWS / 8, 256, 0, stream>>>(ws_x, norm2_w, norm2_b, ws_h);

  // 7) FC1 + exact GELU
  gemm_bf16_kernel<1><<<dim3(MROWS / 128, HID / 64), 256, 0, stream>>>(
      ws_h, wfc1T, fc1_b, ws_mlp, nullptr, nullptr, CC, HID);

  // 8) FC2 + residual -> d_out (f32)
  gemm_bf16_kernel<2><<<dim3(MROWS / 128, CC / 64), 256, 0, stream>>>(
      ws_mlp, wfc2T, fc2_b, nullptr, out, ws_x, HID, CC);
}
